// SelfBlock_90623809946406
// MI455X (gfx1250) — matrix operand order repaired and checked by reading, not compiled, not run on hardware
//
#include <hip/hip_runtime.h>
#include <stdint.h>

// ---------------------------------------------------------------------------
// MI455X (gfx1250) transformer self-block.
// Compute-bound (~103 GFLOP vs ~60MB traffic @ 23.3TB/s) -> everything GEMM-
// like goes through v_wmma_f32_16x16x32_bf16 (f32 accumulate).
// Double-buffered LDS tiles + CDNA5 async global->LDS copies
// (GLOBAL_LOAD_ASYNC_TO_LDS_B128, ASYNCcnt) when the builtin is available.
// ---------------------------------------------------------------------------

typedef __attribute__((ext_vector_type(16))) __bf16 v16bf;
typedef __attribute__((ext_vector_type(8)))  float  v8f;
typedef __attribute__((ext_vector_type(4)))  int    v4i;

#define BDIM   4
#define SEQ    1024
#define DMODEL 1024
#define NHEADS 16
#define HD     64

#define AS1 __attribute__((address_space(1)))
#define AS3 __attribute__((address_space(3)))

#if defined(__has_builtin)
#  if __has_builtin(__builtin_amdgcn_global_load_async_to_lds_b128)
#    define HAVE_ASYNC_LDS 1
#  endif
#endif
#ifndef HAVE_ASYNC_LDS
#  define HAVE_ASYNC_LDS 0
#endif

__device__ __forceinline__ unsigned short f2bf(float f) {
  union { float f; uint32_t u; } c; c.f = f;
  uint32_t u = c.u;
  uint32_t r = u + 0x7FFFu + ((u >> 16) & 1u);   // round-to-nearest-even
  return (unsigned short)(r >> 16);
}

// 16-byte global->LDS copy. Async (ASYNCcnt) on gfx1250 toolchains that
// declare the builtin; else classic load + ds_store. The LDS operand is the
// low 32 bits of the generic address (LDS aperture).
__device__ __forceinline__ void copy16_g2l(const unsigned short* g, unsigned short* l) {
#if HAVE_ASYNC_LDS
  __builtin_amdgcn_global_load_async_to_lds_b128(
      (AS1 v4i*)(uintptr_t)g,
      (AS3 v4i*)(uint32_t)(uintptr_t)l, 0, 0);
#else
  *(uint4*)l = *(const uint4*)g;
#endif
}

__device__ __forceinline__ void g2l_wait() {
#if HAVE_ASYNC_LDS
#  if __has_builtin(__builtin_amdgcn_s_wait_asynccnt)
  __builtin_amdgcn_s_wait_asynccnt(0);
#  else
  asm volatile("s_wait_asynccnt 0" ::: "memory");
#  endif
#endif
}

union AB_frag { v16bf v; uint4 u[2]; };          // 16 bf16 per lane = 8 VGPRs

#define WMMA_BF16(A, B, C) \
  __builtin_amdgcn_wmma_f32_16x16x32_bf16(false, (A), false, (B), (short)0, (C), false, false)

// ---------------------------------------------------------------------------
// fp32 -> bf16 converters
// ---------------------------------------------------------------------------
__global__ void cvt_bf16(const float* __restrict__ src,
                         unsigned short* __restrict__ dst, int n4) {
  int i = blockIdx.x * blockDim.x + threadIdx.x;
  if (i >= n4) return;
  float4 f = ((const float4*)src)[i];
  ushort4 o;
  o.x = f2bf(f.x); o.y = f2bf(f.y); o.z = f2bf(f.z); o.w = f2bf(f.w);
  ((ushort4*)dst)[i] = o;
}

__global__ void cvt_bf16_strided(const float* __restrict__ src,
                                 unsigned short* __restrict__ dst,
                                 int rows, int cols4, int dst_ld) {
  int i = blockIdx.x * blockDim.x + threadIdx.x;
  if (i >= rows * cols4) return;
  int r = i / cols4, c = i - r * cols4;
  float4 f = ((const float4*)src)[i];
  ushort4 o;
  o.x = f2bf(f.x); o.y = f2bf(f.y); o.z = f2bf(f.z); o.w = f2bf(f.w);
  *(ushort4*)&dst[(size_t)r * dst_ld + c * 4] = o;
}

// ---------------------------------------------------------------------------
// GEMM: C[M,N] = A[M,K](bf16,lda) @ W[N,K]^T(bf16,ldw) + bias (+resid)
// 128x128 tile per workgroup, 8 waves, wave subtile 32x64 (2x4 WMMA C tiles),
// K step 32, double-buffered LDS (rows padded to 40 elems: 80B, 16B aligned,
// distinct banks across 16 consecutive rows). Next tile's global->LDS copy is
// issued before compute on the current tile and waited only at step end.
// ---------------------------------------------------------------------------
__global__ __launch_bounds__(256) void gemm_bf16_128x128(
    const unsigned short* __restrict__ A, int lda,
    const unsigned short* __restrict__ W, int ldw,
    const float* __restrict__ bias,
    float* __restrict__ outF, unsigned short* __restrict__ outB,
    int ldo, int ocol0,
    const float* __restrict__ resid,
    int K)
{
  __shared__ unsigned short As[2][128 * 40];
  __shared__ unsigned short Ws[2][128 * 40];

  const int t    = threadIdx.x;
  const int lane = t & 31;
  const int wave = t >> 5;
  const int wr   = wave >> 1;       // 0..3 (M direction)
  const int wc   = wave & 1;        // 0..1 (N direction)
  const int l16  = lane & 15;
  const int hi   = (lane >> 4) & 1;
  const int m0   = blockIdx.y * 128;
  const int n0   = blockIdx.x * 128;

  // per-thread tile-copy coordinates (2 x 16B chunks each for A and W)
  const int row0 = t >> 2,           seg0 = t & 3;
  const int row1 = (t + 256) >> 2,   seg1 = seg0;

  v8f zero = {0,0,0,0,0,0,0,0};
  v8f acc[2][4];
#pragma unroll
  for (int i = 0; i < 2; i++)
#pragma unroll
    for (int j = 0; j < 4; j++) acc[i][j] = zero;

  const int nk = K >> 5;

  // prologue: tile 0 -> buffer 0
  copy16_g2l(&A[(size_t)(m0 + row0) * lda + seg0 * 8], &As[0][row0 * 40 + seg0 * 8]);
  copy16_g2l(&W[(size_t)(n0 + row0) * ldw + seg0 * 8], &Ws[0][row0 * 40 + seg0 * 8]);
  copy16_g2l(&A[(size_t)(m0 + row1) * lda + seg1 * 8], &As[0][row1 * 40 + seg1 * 8]);
  copy16_g2l(&W[(size_t)(n0 + row1) * ldw + seg1 * 8], &Ws[0][row1 * 40 + seg1 * 8]);
  g2l_wait();
  __syncthreads();

  for (int kk = 0; kk < nk; kk++) {
    const int buf = kk & 1;
    if (kk + 1 < nk) {                       // issue next tile, wait later
      int k0 = (kk + 1) << 5;
      copy16_g2l(&A[(size_t)(m0 + row0) * lda + k0 + seg0 * 8], &As[buf ^ 1][row0 * 40 + seg0 * 8]);
      copy16_g2l(&W[(size_t)(n0 + row0) * ldw + k0 + seg0 * 8], &Ws[buf ^ 1][row0 * 40 + seg0 * 8]);
      copy16_g2l(&A[(size_t)(m0 + row1) * lda + k0 + seg1 * 8], &As[buf ^ 1][row1 * 40 + seg1 * 8]);
      copy16_g2l(&W[(size_t)(n0 + row1) * ldw + k0 + seg1 * 8], &Ws[buf ^ 1][row1 * 40 + seg1 * 8]);
    }

    AB_frag a[2], b[4];
#pragma unroll
    for (int mf = 0; mf < 2; mf++) {                    // A: lane=row, K packed
      int ar = wr * 32 + mf * 16 + l16;
      a[mf].u[0] = *(const uint4*)&As[buf][ar * 40 + hi * 8];
      a[mf].u[1] = *(const uint4*)&As[buf][ar * 40 + hi * 8 + 16];
    }
#pragma unroll
    for (int nf = 0; nf < 4; nf++) {                    // B: lane=col, K packed
      int brw = wc * 64 + nf * 16 + l16;
      b[nf].u[0] = *(const uint4*)&Ws[buf][brw * 40 + hi * 8];
      b[nf].u[1] = *(const uint4*)&Ws[buf][brw * 40 + hi * 8 + 16];
    }
#pragma unroll
    for (int mf = 0; mf < 2; mf++)
#pragma unroll
      for (int nf = 0; nf < 4; nf++)
        acc[mf][nf] = WMMA_BF16(a[mf].v, b[nf].v, acc[mf][nf]);

    g2l_wait();                              // next tile fully in LDS
    __syncthreads();
  }

  // epilogue: bias (+resid), f32 or bf16 store (C layout: row=r+8*hi, col=l16)
#pragma unroll
  for (int mf = 0; mf < 2; mf++) {
#pragma unroll
    for (int nf = 0; nf < 4; nf++) {
      int gn = n0 + wc * 64 + nf * 16 + l16;
      float bv = bias ? bias[gn] : 0.0f;
#pragma unroll
      for (int r = 0; r < 8; r++) {
        int gm = m0 + wr * 32 + mf * 16 + r + hi * 8;
        float v = acc[mf][nf][r] + bv;
        if (resid) v += resid[(size_t)gm * ldo + ocol0 + gn];
        if (outF) outF[(size_t)gm * ldo + ocol0 + gn] = v;
        else      outB[(size_t)gm * ldo + ocol0 + gn] = f2bf(v);
      }
    }
  }
}

// ---------------------------------------------------------------------------
// RoPE + qkv split: qkv f32 [B*N, 3D] (inner layout h*192 + d*3 + {q,k,v})
// -> q,k,v bf16 [B,H,N,HD]; attention scale HD^-0.5 folded into q.
// ---------------------------------------------------------------------------
__global__ void rope_split(const float* __restrict__ qkv,
                           const float* __restrict__ enc,
                           unsigned short* __restrict__ Q,
                           unsigned short* __restrict__ Kk,
                           unsigned short* __restrict__ V)
{
  int p  = blockIdx.x * blockDim.x + threadIdx.x;  // B*H*N*32 threads
  int d2 = p & 31;
  int n  = (p >> 5) & (SEQ - 1);
  int h  = (p >> 15) & (NHEADS - 1);
  int b  = p >> 19;
  int d0 = d2 * 2, d1 = d0 + 1;

  size_t rbase = ((size_t)(b * SEQ + n)) * 3072 + h * 192;
  float q0v = qkv[rbase + d0 * 3 + 0];
  float q1v = qkv[rbase + d1 * 3 + 0];
  float k0v = qkv[rbase + d0 * 3 + 1];
  float k1v = qkv[rbase + d1 * 3 + 1];
  float v0v = qkv[rbase + d0 * 3 + 2];
  float v1v = qkv[rbase + d1 * 3 + 2];

  size_t e0 = ((size_t)(b * SEQ + n)) * HD;                 // cos
  size_t e1 = ((size_t)((BDIM + b) * SEQ + n)) * HD;        // sin
  float c0 = enc[e0 + d0], c1 = enc[e0 + d1];
  float s0 = enc[e1 + d0], s1 = enc[e1 + d1];

  // rotate_half: out[2i] = -t[2i+1], out[2i+1] = t[2i]
  float qr0 = q0v * c0 - q1v * s0;
  float qr1 = q1v * c1 + q0v * s1;
  float kr0 = k0v * c0 - k1v * s0;
  float kr1 = k1v * c1 + k0v * s1;

  size_t ob = ((size_t)((b * NHEADS + h) * SEQ + n)) * HD;
  const float sc = 0.125f;  // HD^-0.5
  Q[ob + d0]  = f2bf(qr0 * sc); Q[ob + d1]  = f2bf(qr1 * sc);
  Kk[ob + d0] = f2bf(kr0);      Kk[ob + d1] = f2bf(kr1);
  V[ob + d0]  = f2bf(v0v);      V[ob + d1]  = f2bf(v1v);
}

// ---------------------------------------------------------------------------
// Flash attention: one workgroup = (b,h) x 128-query tile; 8 waves x 16 rows.
// K tile row-major in LDS (stride 72: K^T B-frags contiguous per lane),
// V tile transposed in LDS (stride 136: P.V B-frags contiguous per lane),
// P converted C-layout -> A-layout through per-wave LDS staging (stride 56).
// K tile copied with async global->LDS when available.
// ---------------------------------------------------------------------------
__global__ __launch_bounds__(256) void flash_attn(
    const unsigned short* __restrict__ Q,
    const unsigned short* __restrict__ Kg,
    const unsigned short* __restrict__ Vg,
    unsigned short* __restrict__ ctx)
{
  __shared__ unsigned short Ks[128 * 72];
  __shared__ unsigned short Vt[64 * 136];
  __shared__ unsigned short Ps[8 * 16 * 56];

  const int t    = threadIdx.x;
  const int lane = t & 31;
  const int wave = t >> 5;
  const int l16  = lane & 15;
  const int hi   = (lane >> 4) & 1;
  const int bh   = blockIdx.y;
  const int b    = bh >> 4;
  const int h    = bh & 15;
  const int q0row = blockIdx.x * 128;
  const size_t headOff = (size_t)bh * SEQ * HD;

  // Q A-fragments (dims 0..31 and 32..63), 16 rows per wave
  AB_frag qlo, qhi;
  {
    int row = q0row + wave * 16 + l16;
    const unsigned short* qp = Q + headOff + (size_t)row * HD;
    qlo.u[0] = *(const uint4*)(qp + hi * 8);
    qlo.u[1] = *(const uint4*)(qp + hi * 8 + 16);
    qhi.u[0] = *(const uint4*)(qp + 32 + hi * 8);
    qhi.u[1] = *(const uint4*)(qp + 32 + hi * 8 + 16);
  }

  v8f zero = {0,0,0,0,0,0,0,0};
  float mrow[8], lrow[8];
  v8f ctxa[4];
#pragma unroll
  for (int r = 0; r < 8; r++) { mrow[r] = -3.0e38f; lrow[r] = 0.0f; }
#pragma unroll
  for (int c = 0; c < 4; c++) ctxa[c] = zero;

  unsigned short* myPs = &Ps[wave * 16 * 56];

  for (int kt = 0; kt < SEQ; kt += 128) {
    __syncthreads();
    // K tile 128x64 row-major (async copy path when available)
#pragma unroll
    for (int i = 0; i < 4; i++) {
      int c = t + i * 256;
      int row = c >> 3, seg = c & 7;
      copy16_g2l(&Kg[headOff + (size_t)(kt + row) * HD + seg * 8],
                 &Ks[row * 72 + seg * 8]);
    }
    // V tile transposed: Vt[dim][key] (per-element, stays on the sync path)
    {
      int key = t >> 1, half = t & 1;
      union { uint4 q[4]; unsigned short s[32]; } buf;
      const unsigned short* vp = &Vg[headOff + (size_t)(kt + key) * HD + half * 32];
#pragma unroll
      for (int i = 0; i < 4; i++) buf.q[i] = *(const uint4*)(vp + i * 8);
#pragma unroll
      for (int j = 0; j < 32; j++) Vt[(half * 32 + j) * 136 + key] = buf.s[j];
    }
    g2l_wait();
    __syncthreads();

#pragma unroll
    for (int ch = 0; ch < 4; ch++) {          // 32 keys per chunk
      int j0 = ch * 32;
      v8f s0t = zero, s1t = zero;
      {
        AB_frag ka, kb2;
        int key = j0 + l16;                   // keys j0..j0+15
        ka.u[0]  = *(const uint4*)&Ks[key * 72 + hi * 8];
        ka.u[1]  = *(const uint4*)&Ks[key * 72 + hi * 8 + 16];
        kb2.u[0] = *(const uint4*)&Ks[key * 72 + 32 + hi * 8];
        kb2.u[1] = *(const uint4*)&Ks[key * 72 + 32 + hi * 8 + 16];
        s0t = WMMA_BF16(qlo.v, ka.v,  s0t);
        s0t = WMMA_BF16(qhi.v, kb2.v, s0t);
        key = j0 + 16 + l16;                  // keys j0+16..j0+31
        ka.u[0]  = *(const uint4*)&Ks[key * 72 + hi * 8];
        ka.u[1]  = *(const uint4*)&Ks[key * 72 + hi * 8 + 16];
        kb2.u[0] = *(const uint4*)&Ks[key * 72 + 32 + hi * 8];
        kb2.u[1] = *(const uint4*)&Ks[key * 72 + 32 + hi * 8 + 16];
        s1t = WMMA_BF16(qlo.v, ka.v,  s1t);
        s1t = WMMA_BF16(qhi.v, kb2.v, s1t);
      }

      // online softmax (row = C-layout: VGPR r + 8*hi; 16-lane reductions)
      float p0[8], p1[8], alpha[8];
#pragma unroll
      for (int r = 0; r < 8; r++) {
        float mx = fmaxf(s0t[r], s1t[r]);
        mx = fmaxf(mx, __shfl_xor(mx, 1, 32));
        mx = fmaxf(mx, __shfl_xor(mx, 2, 32));
        mx = fmaxf(mx, __shfl_xor(mx, 4, 32));
        mx = fmaxf(mx, __shfl_xor(mx, 8, 32));
        float mnew = fmaxf(mrow[r], mx);
        float al = __expf(mrow[r] - mnew);
        mrow[r] = mnew;
        float e0 = __expf(s0t[r] - mnew);
        float e1 = __expf(s1t[r] - mnew);
        p0[r] = e0; p1[r] = e1;
        float rs = e0 + e1;
        rs += __shfl_xor(rs, 1, 32);
        rs += __shfl_xor(rs, 2, 32);
        rs += __shfl_xor(rs, 4, 32);
        rs += __shfl_xor(rs, 8, 32);
        lrow[r] = lrow[r] * al + rs;
        alpha[r] = al;
      }
#pragma unroll
      for (int c = 0; c < 4; c++)
#pragma unroll
        for (int r = 0; r < 8; r++) ctxa[c][r] *= alpha[r];

      // P: C layout -> LDS -> A layout (LDS ops are in-order within a wave)
#pragma unroll
      for (int r = 0; r < 8; r++) {
        int prow = r + hi * 8;
        myPs[prow * 56 + l16]      = f2bf(p0[r]);
        myPs[prow * 56 + 16 + l16] = f2bf(p1[r]);
      }
      asm volatile("" ::: "memory");
      __builtin_amdgcn_wave_barrier();
      AB_frag pf;
      pf.u[0] = *(const uint4*)&myPs[l16 * 56 + hi * 8];
      pf.u[1] = *(const uint4*)&myPs[l16 * 56 + hi * 8 + 16];

#pragma unroll
      for (int c = 0; c < 4; c++) {
        AB_frag vf;
        int dim = c * 16 + l16;
        vf.u[0] = *(const uint4*)&Vt[dim * 136 + j0 + hi * 8];
        vf.u[1] = *(const uint4*)&Vt[dim * 136 + j0 + hi * 8 + 16];
        ctxa[c] = WMMA_BF16(pf.v, vf.v, ctxa[c]);
      }
    }
  }

  // finalize: ctx / l, store bf16 to [B*N, D] with col = h*64 + d
#pragma unroll
  for (int r = 0; r < 8; r++) lrow[r] = 1.0f / lrow[r];
  int orow0 = b * SEQ + q0row + wave * 16;
#pragma unroll
  for (int c = 0; c < 4; c++)
#pragma unroll
    for (int r = 0; r < 8; r++) {
      int grow = orow0 + r + hi * 8;
      int gcol = h * HD + c * 16 + l16;
      ctx[(size_t)grow * DMODEL + gcol] = f2bf(ctxa[c][r] * lrow[r]);
    }
}

// ---------------------------------------------------------------------------
// Row LayerNorm (2048) + exact GELU -> bf16
// ---------------------------------------------------------------------------
__global__ __launch_bounds__(256) void ln_gelu(
    const float* __restrict__ hsrc, const float* __restrict__ g,
    const float* __restrict__ bta, unsigned short* __restrict__ dst)
{
  __shared__ float red[256];
  int row = blockIdx.x;
  int t = threadIdx.x;
  const float* hp = hsrc + (size_t)row * 2048;
  float loc[8];
  float s = 0.0f;
#pragma unroll
  for (int i = 0; i < 8; i++) { loc[i] = hp[t + i * 256]; s += loc[i]; }
  red[t] = s; __syncthreads();
  for (int off = 128; off > 0; off >>= 1) {
    if (t < off) red[t] += red[t + off];
    __syncthreads();
  }
  float mu = red[0] * (1.0f / 2048.0f);
  __syncthreads();
  float vs = 0.0f;
#pragma unroll
  for (int i = 0; i < 8; i++) { float d = loc[i] - mu; vs += d * d; }
  red[t] = vs; __syncthreads();
  for (int off = 128; off > 0; off >>= 1) {
    if (t < off) red[t] += red[t + off];
    __syncthreads();
  }
  float rstd = rsqrtf(red[0] * (1.0f / 2048.0f) + 1e-5f);
#pragma unroll
  for (int i = 0; i < 8; i++) {
    int col = t + i * 256;
    float xv = (loc[i] - mu) * rstd * g[col] + bta[col];
    float y = 0.5f * xv * (1.0f + erff(xv * 0.70710678118f));
    dst[(size_t)row * 2048 + col] = f2bf(y);
  }
}

// ---------------------------------------------------------------------------
extern "C" void kernel_launch(void* const* d_in, const int* in_sizes, int n_in,
                              void* d_out, int out_size, void* d_ws, size_t ws_size,
                              hipStream_t stream) {
  const float* x      = (const float*)d_in[0];
  const float* enc    = (const float*)d_in[1];
  const float* Wqkv_w = (const float*)d_in[2];
  const float* Wqkv_b = (const float*)d_in[3];
  const float* out_w  = (const float*)d_in[4];
  const float* out_b  = (const float*)d_in[5];
  const float* ffn1_w = (const float*)d_in[6];
  const float* ffn1_b = (const float*)d_in[7];
  const float* ln_g   = (const float*)d_in[8];
  const float* ln_b   = (const float*)d_in[9];
  const float* ffn2_w = (const float*)d_in[10];
  const float* ffn2_b = (const float*)d_in[11];
  float* out = (float*)d_out;

  char* ws = (char*)d_ws;
  size_t off = 0;
  auto alloc = [&](size_t bytes) -> void* {
    void* p = ws + off;
    off += (bytes + 255) & ~(size_t)255;
    return p;
  };
  float*          qkv   = (float*)alloc(4096ULL * 3072 * 4);   // reused as FFN1 f32 out
  unsigned short* xc    = (unsigned short*)alloc(4096ULL * 2048 * 2);
  unsigned short* qb    = (unsigned short*)alloc(64ULL * 1024 * 64 * 2);
  unsigned short* kb    = (unsigned short*)alloc(64ULL * 1024 * 64 * 2);
  unsigned short* vb    = (unsigned short*)alloc(64ULL * 1024 * 64 * 2);
  unsigned short* ctxb  = (unsigned short*)alloc(4096ULL * 1024 * 2);
  unsigned short* hb    = (unsigned short*)alloc(4096ULL * 2048 * 2);
  unsigned short* wqkvB = (unsigned short*)alloc(3072ULL * 1024 * 2);
  unsigned short* outwB = (unsigned short*)alloc(1024ULL * 1024 * 2);
  unsigned short* f1B   = (unsigned short*)alloc(2048ULL * 2048 * 2);
  unsigned short* f2B   = (unsigned short*)alloc(1024ULL * 2048 * 2);
  float* hbuf = qkv;  // alias: qkv dead after rope_split

  // weights -> bf16
  cvt_bf16<<<3072 * 1024 / 4 / 256, 256, 0, stream>>>(Wqkv_w, wqkvB, 3072 * 1024 / 4);
  cvt_bf16<<<1024 * 1024 / 4 / 256, 256, 0, stream>>>(out_w,  outwB, 1024 * 1024 / 4);
  cvt_bf16<<<2048 * 2048 / 4 / 256, 256, 0, stream>>>(ffn1_w, f1B,   2048 * 2048 / 4);
  cvt_bf16<<<2048 * 1024 / 4 / 256, 256, 0, stream>>>(ffn2_w, f2B,   2048 * 1024 / 4);
  // x -> xc[:, 0:1024]
  cvt_bf16_strided<<<4096 * 256 / 256, 256, 0, stream>>>(x, xc, 4096, 256, 2048);

  // QKV: [4096,1024] @ [3072,1024]^T -> f32 [4096,3072]
  gemm_bf16_128x128<<<dim3(24, 32), 256, 0, stream>>>(
      xc, 2048, wqkvB, 1024, Wqkv_b, qkv, nullptr, 3072, 0, nullptr, 1024);

  // RoPE + split -> q,k,v bf16 [B,H,N,64]
  rope_split<<<(BDIM * NHEADS * SEQ * 32) / 256, 256, 0, stream>>>(qkv, enc, qb, kb, vb);

  // Flash attention -> ctx bf16 [B*N, 1024]
  flash_attn<<<dim3(SEQ / 128, BDIM * NHEADS), 256, 0, stream>>>(qb, kb, vb, ctxb);

  // message = ctx @ out_w^T + out_b -> bf16 into xc[:, 1024:2048]
  gemm_bf16_128x128<<<dim3(8, 32), 256, 0, stream>>>(
      ctxb, 1024, outwB, 1024, out_b, nullptr, xc, 2048, 1024, nullptr, 1024);

  // FFN1: xc[4096,2048] @ ffn1_w^T -> f32 [4096,2048]
  gemm_bf16_128x128<<<dim3(16, 32), 256, 0, stream>>>(
      xc, 2048, f1B, 2048, ffn1_b, hbuf, nullptr, 2048, 0, nullptr, 2048);

  // LayerNorm + exact GELU -> bf16
  ln_gelu<<<4096, 256, 0, stream>>>(hbuf, ln_g, ln_b, hb);

  // FFN2 + bias + residual(x) -> f32 d_out
  gemm_bf16_128x128<<<dim3(8, 32), 256, 0, stream>>>(
      hb, 2048, f2B, 2048, ffn2_b, out, nullptr, 1024, 0, x, 2048);
}
